// VaderRNN_1116691497366
// MI455X (gfx1250) — matrix operand order repaired and checked by reading, not compiled, not run on hardware
//
#include <hip/hip_runtime.h>
#include <hip/hip_bf16.h>
#include <math.h>

// ---------------------------------------------------------------------------
// Problem constants
// ---------------------------------------------------------------------------
#define BB   512   // batch
#define TT   128   // time steps (D)
#define II   64    // input dim
#define HH   256   // decoder hidden
#define HE   512   // encoder hidden
#define NTHREADS 512   // 16 waves per workgroup (wave32)

typedef __bf16 bf16_t;
typedef __attribute__((ext_vector_type(16))) __bf16 bfx16;
typedef __attribute__((ext_vector_type(8)))  __bf16 bfx8;
typedef __attribute__((ext_vector_type(8)))  float  fx8;

// ---------------------------------------------------------------------------
// WMMA helpers (CDNA5: wave32, 16x16x32 bf16 -> f32)
// ---------------------------------------------------------------------------
__device__ __forceinline__ fx8 wmma_bf16(bfx16 a, bfx16 b, fx8 c) {
  return __builtin_amdgcn_wmma_f32_16x16x32_bf16(
      /*neg_a=*/false, a, /*neg_b=*/false, b,
      /*c_mod=*/(short)0, c, /*reuse_a=*/false, /*reuse_b=*/false);
}

// A fragment (16x32 bf16, row-major source): lane<16 holds M=lane,
// K = {0..7} u {16..23}; lane>=16 holds M=lane-16, K = {8..15} u {24..31}.
// `row` points at this lane's M-row; koff = kk*32 + (lane>=16 ? 8 : 0).
__device__ __forceinline__ bfx16 load_a_frag(const bf16_t* row, int koff) {
  union { bfx16 v; bfx8 h[2]; } u;
  u.h[0] = *(const bfx8*)(row + koff);
  u.h[1] = *(const bfx8*)(row + koff + 16);
  return u.v;
}

__device__ __forceinline__ float sigmoidf_(float x) {
  return 1.0f / (1.0f + __expf(-x));
}

// ---------------------------------------------------------------------------
// Weight pre-pack: row-major fp32 [Ktot, Ntot] -> bf16 WMMA-B fragments.
// packed[((nn*KC + kk)*32 + lane)*16 + e] = W[K, N]
//   N = nn*16 + (lane&15);  K = kk*32 + (lane>=16 ? 8 : 0) + (e&7) + (e>=8 ? 16 : 0)
// Each lane's 16 bf16 (32B) are contiguous -> pure b128 streaming loads later.
// ---------------------------------------------------------------------------
__global__ void pack_b_kernel(const float* __restrict__ src, bf16_t* __restrict__ dst,
                              int Ktot, int Ntot) {
  int idx = blockIdx.x * 256 + threadIdx.x;
  int total = Ktot * Ntot;
  if (idx >= total) return;
  int e    = idx & 15;
  int lane = (idx >> 4) & 31;
  int rest = idx >> 9;
  int KC   = Ktot >> 5;
  int kk   = rest % KC;
  int nn   = rest / KC;
  int N = nn * 16 + (lane & 15);
  int K = kk * 32 + ((lane >= 16) ? 8 : 0) + (e & 7) + ((e & 8) ? 16 : 0);
  dst[idx] = (bf16_t)src[K * Ntot + N];
}

// ---------------------------------------------------------------------------
// Encoder: peephole LSTM, I=64 -> He=512, fused imputation.
// Grid: 32 WGs (16 batch rows each), 512 threads (16 waves).
// Wave w owns hidden units [w*32, w*32+32) for ALL four gates:
//   n-tile(g,j) = g*32 + w*2 + j, j in 0..1.  Cell state c lives in registers
//   in accumulator layout (acc[4][2] = 64 VGPRs, creg[2] = 16 VGPRs).
//   h is staged in LDS as bf16 for the next step's h@Wh.
// ---------------------------------------------------------------------------
__global__ __launch_bounds__(NTHREADS) void encoder_kernel(
    const float* __restrict__ X, const float* __restrict__ Wm,
    const float* __restrict__ Aimp, const float* __restrict__ b_e,
    const float* __restrict__ wci, const float* __restrict__ wcf,
    const float* __restrict__ wco, const float* __restrict__ noise,
    const bf16_t* __restrict__ packWx, const bf16_t* __restrict__ packWh,
    float* __restrict__ out_z, float* __restrict__ out_mu,
    float* __restrict__ out_ls)
{
  __shared__ bf16_t x_lds[16 * II];   // 2 KB  (imputed input tile, bf16)
  __shared__ bf16_t h_bf[16 * HE];    // 16 KB (h_t, bf16, A-source for h@Wh)
  __shared__ float  h_f32[16 * HE];   // 32 KB (final h_T, fp32)

  const int tid  = threadIdx.x;
  const int lane = tid & 31;
  const int wv   = tid >> 5;          // 0..15
  const int li   = lane & 15;
  const int hi8  = (lane >> 4) & 1;
  const int b0   = blockIdx.x * 16;
  const int abase = hi8 ? 8 : 0;

  for (int s = tid; s < 16 * HE; s += NTHREADS) h_bf[s] = (bf16_t)0.0f;

  // step-invariant per-lane params
  float bval[4][2], wciv[2], wcfv[2], wcov[2];
#pragma unroll
  for (int j = 0; j < 2; ++j) {
    int hu = wv * 32 + j * 16 + li;
    wciv[j] = wci[hu]; wcfv[j] = wcf[hu]; wcov[j] = wco[hu];
#pragma unroll
    for (int g = 0; g < 4; ++g) bval[g][j] = b_e[g * HE + hu];
  }

  fx8 creg[2];
#pragma unroll
  for (int j = 0; j < 2; ++j)
#pragma unroll
    for (int q = 0; q < 8; ++q) creg[j][q] = 0.0f;

  const bf16_t* xrow = x_lds + li * II;
  const bf16_t* hrow = h_bf + li * HE;

  for (int t = 0; t < TT; ++t) {
    // ---- fused imputation: X*W + A*(1-W), to bf16 tile in LDS ----
    for (int s = tid; s < 16 * II; s += NTHREADS) {
      int m = s >> 6, i = s & 63;
      int gi = ((b0 + m) * TT + t) * II + i;
      float w  = Wm[gi];
      float xi = X[gi] * w + Aimp[t * II + i] * (1.0f - w);
      x_lds[s] = (bf16_t)xi;
    }
    __syncthreads();

    // ---- z = x@Wx + h@Wh + b  (bf16 WMMA, f32 accum) ----
    fx8 acc[4][2];
#pragma unroll
    for (int g = 0; g < 4; ++g)
#pragma unroll
      for (int j = 0; j < 2; ++j)
#pragma unroll
        for (int q = 0; q < 8; ++q) acc[g][j][q] = bval[g][j];

#pragma unroll 1
    for (int kk = 0; kk < 2; ++kk) {              // x part: K=64
      bfx16 a = load_a_frag(xrow, kk * 32 + abase);
#pragma unroll
      for (int g = 0; g < 4; ++g)
#pragma unroll
        for (int j = 0; j < 2; ++j) {
          int tn = g * 32 + wv * 2 + j;
          bfx16 b = *(const bfx16*)(packWx + ((tn * 2 + kk) * 32 + lane) * 16);
          acc[g][j] = wmma_bf16(a, b, acc[g][j]);
        }
    }
#pragma unroll 1
    for (int kk = 0; kk < 16; ++kk) {             // h part: K=512
      bfx16 a = load_a_frag(hrow, kk * 32 + abase);
#pragma unroll
      for (int g = 0; g < 4; ++g)
#pragma unroll
        for (int j = 0; j < 2; ++j) {
          int tn = g * 32 + wv * 2 + j;
          bfx16 b = *(const bfx16*)(packWh + ((tn * 16 + kk) * 32 + lane) * 16);
          acc[g][j] = wmma_bf16(a, b, acc[g][j]);
        }
    }
    __syncthreads();

    // ---- peephole gates (wave-local; c in registers, acc layout) ----
#pragma unroll
    for (int j = 0; j < 2; ++j) {
      int hu = wv * 32 + j * 16 + li;
#pragma unroll
      for (int v = 0; v < 8; ++v) {
        float c  = creg[j][v];
        float ig = sigmoidf_(acc[0][j][v] + c * wciv[j]);
        float fg = sigmoidf_(acc[1][j][v] + c * wcfv[j]);
        float cn = fg * c + ig * tanhf(acc[2][j][v]);
        float og = sigmoidf_(acc[3][j][v] + cn * wcov[j]);
        float hn = og * tanhf(cn);
        creg[j][v] = cn;
        int m = v + hi8 * 8;
        h_bf[m * HE + hu] = (bf16_t)hn;
        if (t == TT - 1) h_f32[m * HE + hu] = hn;
      }
    }
    __syncthreads();
  }

  // ---- mu_tilde / log_sigma2_tilde / z = mu + exp(0.5 ls)*noise ----
  for (int s = tid; s < 16 * HH; s += NTHREADS) {
    int m = s >> 8, hh = s & 255;
    int b = b0 + m;
    float mu = h_f32[m * HE + hh];
    float ls = h_f32[m * HE + HH + hh];
    float zz = mu + __expf(0.5f * ls) * noise[b * HH + hh];
    out_mu[b * HH + hh] = mu;
    out_ls[b * HH + hh] = ls;
    out_z[b * HH + hh]  = zz;
  }
}

// ---------------------------------------------------------------------------
// Decoder: zero-input peephole LSTM (h0=0, c0=z), H=256, fused per-step
// output projection h_t @ Wout + bout -> sigmoid.
// 512 threads (16 waves); wave w owns hidden units [w*16, w*16+16):
//   n-tile(g) = g*16 + w.  acc[4] = 32 VGPRs, creg = 8 VGPRs.
// Projection (64 cols = 4 tiles) handled by waves 0..3.
// ---------------------------------------------------------------------------
__global__ __launch_bounds__(NTHREADS) void decoder_kernel(
    const float* __restrict__ b_d,
    const float* __restrict__ wci, const float* __restrict__ wcf,
    const float* __restrict__ wco, const float* __restrict__ bout,
    const bf16_t* __restrict__ packWh, const bf16_t* __restrict__ packWo,
    const float* __restrict__ z_in,
    float* __restrict__ out_x, float* __restrict__ out_xraw)
{
  __shared__ bf16_t h_bf[16 * HH];    // 8 KB

  const int tid  = threadIdx.x;
  const int lane = tid & 31;
  const int wv   = tid >> 5;          // 0..15
  const int li   = lane & 15;
  const int hi8  = (lane >> 4) & 1;
  const int b0   = blockIdx.x * 16;
  const int abase = hi8 ? 8 : 0;

  for (int s = tid; s < 16 * HH; s += NTHREADS) h_bf[s] = (bf16_t)0.0f;

  const int hu = wv * 16 + li;
  float bval[4];
#pragma unroll
  for (int g = 0; g < 4; ++g) bval[g] = b_d[g * HH + hu];
  const float wciv = wci[hu], wcfv = wcf[hu], wcov = wco[hu];

  fx8 creg;                           // c0 = z  (acc layout)
#pragma unroll
  for (int v = 0; v < 8; ++v) {
    int m = v + hi8 * 8;
    creg[v] = z_in[(b0 + m) * HH + hu];
  }
  int oc = (wv < 4) ? (wv * 16 + li) : 0;
  float bov = bout[oc];

  const bf16_t* hrow = h_bf + li * HH;

  for (int t = 0; t < TT; ++t) {
    __syncthreads();                  // h_bf stable (init / prev gates+proj)

    // ---- z_pre = h@Wh + b  (dec_in == 0 -> no x term) ----
    fx8 acc[4];
#pragma unroll
    for (int g = 0; g < 4; ++g)
#pragma unroll
      for (int q = 0; q < 8; ++q) acc[g][q] = bval[g];

#pragma unroll 2
    for (int kk = 0; kk < 8; ++kk) {  // K = 256
      bfx16 a = load_a_frag(hrow, kk * 32 + abase);
#pragma unroll
      for (int g = 0; g < 4; ++g) {
        int tn = g * 16 + wv;
        bfx16 b = *(const bfx16*)(packWh + ((tn * 8 + kk) * 32 + lane) * 16);
        acc[g] = wmma_bf16(a, b, acc[g]);
      }
    }
    __syncthreads();

    // ---- gates ----
#pragma unroll
    for (int v = 0; v < 8; ++v) {
      float c  = creg[v];
      float ig = sigmoidf_(acc[0][v] + c * wciv);
      float fg = sigmoidf_(acc[1][v] + c * wcfv);
      float cn = fg * c + ig * tanhf(acc[2][v]);
      float og = sigmoidf_(acc[3][v] + cn * wcov);
      float hn = og * tanhf(cn);
      creg[v] = cn;
      int m = v + hi8 * 8;
      h_bf[m * HH + hu] = (bf16_t)hn;
    }
    __syncthreads();

    // ---- fused output projection: x_raw = h_t @ Wout + bout ----
    if (wv < 4) {                     // wave-uniform branch: EXEC all-1s
      fx8 pacc;
#pragma unroll
      for (int q = 0; q < 8; ++q) pacc[q] = bov;
#pragma unroll 2
      for (int kk = 0; kk < 8; ++kk) {
        bfx16 a = load_a_frag(hrow, kk * 32 + abase);
        bfx16 b = *(const bfx16*)(packWo + ((wv * 8 + kk) * 32 + lane) * 16);
        pacc = wmma_bf16(a, b, pacc);
      }
#pragma unroll
      for (int v = 0; v < 8; ++v) {
        int m   = v + hi8 * 8;
        int idx = ((b0 + m) * TT + t) * II + wv * 16 + li;
        float xr = pacc[v];
        out_xraw[idx] = xr;
        out_x[idx]    = sigmoidf_(xr);
      }
    }
  }
}

// ---------------------------------------------------------------------------
// GMM prior head: mu_c copy, softplus(sigma2), softmax(phi) (K=8).
// ---------------------------------------------------------------------------
__global__ void misc_kernel(const float* __restrict__ mu_c_raw,
                            const float* __restrict__ sig_lat,
                            const float* __restrict__ phi_lat,
                            float* __restrict__ o_muc, float* __restrict__ o_sig,
                            float* __restrict__ o_phi) {
  int tid = blockIdx.x * 256 + threadIdx.x;
  if (tid < 8 * HH) {
    o_muc[tid] = mu_c_raw[tid];
    float v = sig_lat[tid];
    o_sig[tid] = fmaxf(v, 0.0f) + log1pf(__expf(-fabsf(v)));  // softplus
  }
  if (tid == 0) {
    float m = phi_lat[0];
    for (int k = 1; k < 8; ++k) m = fmaxf(m, phi_lat[k]);
    float e[8], s = 0.0f;
    for (int k = 0; k < 8; ++k) { e[k] = __expf(phi_lat[k] - m); s += e[k]; }
    float inv = 1.0f / s;
    for (int k = 0; k < 8; ++k) o_phi[k] = e[k] * inv;
  }
}

// ---------------------------------------------------------------------------
// Launch
// ---------------------------------------------------------------------------
extern "C" void kernel_launch(void* const* d_in, const int* in_sizes, int n_in,
                              void* d_out, int out_size, void* d_ws, size_t ws_size,
                              hipStream_t stream) {
  (void)in_sizes; (void)n_in; (void)out_size; (void)ws_size;

  const float* X     = (const float*)d_in[0];
  const float* Wm    = (const float*)d_in[1];
  const float* noise = (const float*)d_in[2];
  const float* Aimp  = (const float*)d_in[3];
  const float* Wx_e  = (const float*)d_in[4];
  const float* Wh_e  = (const float*)d_in[5];
  const float* b_e   = (const float*)d_in[6];
  const float* wci_e = (const float*)d_in[7];
  const float* wcf_e = (const float*)d_in[8];
  const float* wco_e = (const float*)d_in[9];
  /* d_in[10] = Wx_d: decoder input is identically zero -> unused */
  const float* Wh_d  = (const float*)d_in[11];
  const float* b_d   = (const float*)d_in[12];
  const float* wci_d = (const float*)d_in[13];
  const float* wcf_d = (const float*)d_in[14];
  const float* wco_d = (const float*)d_in[15];
  const float* Wout  = (const float*)d_in[16];
  const float* bout  = (const float*)d_in[17];
  const float* mu_c  = (const float*)d_in[18];
  const float* s_lat = (const float*)d_in[19];
  const float* p_lat = (const float*)d_in[20];

  // Output layout (flat f32, reference tuple order)
  float* out    = (float*)d_out;
  float* o_x    = out;                      // [B,D,I]  4,194,304
  float* o_xraw = out + 4194304;            // [B,D,I]
  float* o_muc  = out + 8388608;            // [K,H]    2048
  float* o_sig  = out + 8390656;            // [K,H]
  float* o_phi  = out + 8392704;            // [K]      8
  float* o_z    = out + 8392712;            // [B,H]    131072
  float* o_mu   = out + 8523784;            // [B,H]
  float* o_ls   = out + 8654856;            // [B,H]

  // Workspace: packed bf16 WMMA-B weights (~2.8 MB, L2-resident)
  bf16_t* wsb  = (bf16_t*)d_ws;
  bf16_t* pWx  = wsb;                                   //  64x2048 = 131072
  bf16_t* pWhE = wsb + 131072;                          // 512x2048 = 1048576
  bf16_t* pWhD = wsb + 131072 + 1048576;                // 256x1024 = 262144
  bf16_t* pWo  = wsb + 131072 + 1048576 + 262144;       // 256x64   = 16384

  pack_b_kernel<<<(64 * 2048 + 255) / 256, 256, 0, stream>>>(Wx_e, pWx, 64, 2048);
  pack_b_kernel<<<(512 * 2048 + 255) / 256, 256, 0, stream>>>(Wh_e, pWhE, 512, 2048);
  pack_b_kernel<<<(256 * 1024 + 255) / 256, 256, 0, stream>>>(Wh_d, pWhD, 256, 1024);
  pack_b_kernel<<<(256 * 64 + 255) / 256, 256, 0, stream>>>(Wout, pWo, 256, 64);

  misc_kernel<<<8, 256, 0, stream>>>(mu_c, s_lat, p_lat, o_muc, o_sig, o_phi);

  encoder_kernel<<<BB / 16, NTHREADS, 0, stream>>>(
      X, Wm, Aimp, b_e, wci_e, wcf_e, wco_e, noise, pWx, pWhE, o_z, o_mu, o_ls);

  decoder_kernel<<<BB / 16, NTHREADS, 0, stream>>>(
      b_d, wci_d, wcf_d, wco_d, bout, pWhD, pWo, o_z, o_x, o_xraw);
}
